// BidirVimRWKV6Encoder_76003741270344
// MI455X (gfx1250) — matrix operand order repaired and checked by reading, not compiled, not run on hardware
//
#include <hip/hip_runtime.h>
#include <cstdint>
#include <cstddef>

// ---------------------------------------------------------------------------
// MI455X (gfx1250) RWKV6 bidirectional encoder.
// GEMMs: v_wmma_f32_16x16x32_bf16 (bf16 A/B, fp32 accum), A tiles staged via
// global_load_async_to_lds_b128 (ASYNCcnt), B tiles transposed through VGPRs.
// ---------------------------------------------------------------------------

typedef unsigned short u16;
typedef unsigned int   u32;
typedef unsigned long long u64;
typedef __attribute__((ext_vector_type(16))) __bf16 v16bf;
typedef __attribute__((ext_vector_type(8)))  float  v8f;
typedef __attribute__((ext_vector_type(8)))  u32    v8u;

enum { OP_NONE = 0, OP_TANH = 1, OP_SILU = 2, OP_SIGMOID = 3, OP_RELUSQ = 4 };
enum { EM_F32 = 0, EM_ACC = 1, EM_BF16 = 2 };

__device__ __forceinline__ u16 f2bf(float f) {
  u32 u = __float_as_uint(f);
  u += 0x7fffu + ((u >> 16) & 1u);       // round-to-nearest-even
  return (u16)(u >> 16);
}

__device__ __forceinline__ v16bf as_v16bf(v8u x) {
  union { v8u u; v16bf v; } c; c.u = x; return c.v;
}

template <int OP>
__device__ __forceinline__ float apply_op(float v) {
  if (OP == OP_TANH)    return tanhf(v);
  if (OP == OP_SILU)    return v / (1.f + expf(-v));
  if (OP == OP_SIGMOID) return 1.f / (1.f + expf(-v));
  if (OP == OP_RELUSQ)  { float r = fmaxf(v, 0.f); return r * r; }
  return v;
}

// ---------------------------------------------------------------------------
// WMMA GEMM: out[M,Nn] = op(A_bf16[M,K] @ B_bf16[K,Nn])
// Block tile 64x64, 8 waves, K-step 32. M,Nn multiples of 64; K multiple of 32.
// ---------------------------------------------------------------------------
#define LDSS 40   // LDS row stride in shorts: 80B rows -> 16B-aligned rows

template <int OP, int EM>
__global__ __launch_bounds__(256) void gemm_bf16_kernel(
    const u16* __restrict__ A, const u16* __restrict__ Bw,
    float* __restrict__ D, u16* __restrict__ Dh,
    int M, int Nn, int K, int lda, int ldb, int ldd)
{
  // single shared block: As at element 0, Bs right after
  __shared__ alignas(16) u16 smem[2 * 64 * LDSS];
  u16* As = smem;                 // [64 rows][32 k] stride LDSS
  u16* Bs = smem + 64 * LDSS;     // transposed [64 n][32 k] stride LDSS

  const int tid  = threadIdx.x;
  const int lane = tid & 31;
  const int wv   = tid >> 5;
  const int row0 = blockIdx.y << 6;
  const int col0 = blockIdx.x << 6;
  const int rw   = (wv & 3) << 4;   // wave row offset in block tile
  const int cw   = (wv >> 2) << 5;  // wave col offset in block tile

  const v8f zero = {0.f, 0.f, 0.f, 0.f, 0.f, 0.f, 0.f, 0.f};
  v8f acc0 = zero, acc1 = zero;

  // staging maps
  const int arow  = tid >> 2;        // 0..63 A row
  const int akseg = (tid & 3) << 3;  // 0,8,16,24 A k-segment (8 bf16 = 16B)
  const int bkrow = tid >> 3;        // 0..31 B k-row
  const int bns   = (tid & 7) << 3;  // 0..56 B n-segment

  // fragment maps (16-bit WMMA; A and B both use contiguous k = 16*half + s)
  const int m    = lane & 15;
  const int half = lane >> 4;
  const int ka   = half << 4;                 // A: K base (0 or 16)
  const int kb   = (half << 4) + 8;           // A: K base + 8
  const int bn   = lane & 15;                 // B/D column within 16
  const int bk0  = (lane >> 4) << 4;          // B: K base (0 or 16)

  // per-thread async A destination: low 32 bits of generic ptr == LDS byte addr
  const u32 ldsA = (u32)(uintptr_t)(void*)&As[arow * LDSS + akseg];

  for (int k0 = 0; k0 < K; k0 += 32) {
    { // stage A tile (64 x 32 bf16 = 4KB) via async DMA to LDS (16B/lane)
      u64 ga = (u64)(uintptr_t)(const void*)
               (A + (size_t)(row0 + arow) * lda + (k0 + akseg));
      asm volatile("global_load_async_to_lds_b128 %0, %1, off"
                   :: "v"(ldsA), "v"(ga) : "memory");
    }
    { // stage B tile transposed (cols -> rows)
      const u16* bg = Bw + (size_t)(k0 + bkrow) * ldb + (col0 + bns);
      uint4 v = *(const uint4*)bg;
      u32 w0[4] = {v.x, v.y, v.z, v.w};
      u16 tmp[8];
      #pragma unroll
      for (int e = 0; e < 4; e++) {
        tmp[2 * e]     = (u16)(w0[e] & 0xffffu);
        tmp[2 * e + 1] = (u16)(w0[e] >> 16);
      }
      #pragma unroll
      for (int e = 0; e < 8; e++) Bs[(bns + e) * LDSS + bkrow] = tmp[e];
    }
    if (k0 + 32 < K)  // gfx1250 global_prefetch_b8 for next B tile
      __builtin_prefetch(Bw + (size_t)(k0 + 32 + bkrow) * ldb + (col0 + bns), 0, 1);

    asm volatile("s_wait_asynccnt 0x0" ::: "memory");  // my async A landed
    __syncthreads();                                   // everyone's tiles ready

    // A fragment: lane(m,half): V0..7 <- K=16h..16h+15
    v8u au;
    const u16* arp = &As[(rw + m) * LDSS];
    #pragma unroll
    for (int p = 0; p < 4; p++) au[p]     = *(const u32*)&arp[ka + 2 * p];
    #pragma unroll
    for (int p = 0; p < 4; p++) au[4 + p] = *(const u32*)&arp[kb + 2 * p];
    v16bf av = as_v16bf(au);

    // B fragments: lane(n,halfK): Vp packs K = bk0+2p, bk0+2p+1 at column n
    v8u bu0, bu1;
    const u16* br0 = &Bs[(cw + bn) * LDSS + bk0];
    const u16* br1 = &Bs[(cw + 16 + bn) * LDSS + bk0];
    #pragma unroll
    for (int p = 0; p < 8; p++) bu0[p] = *(const u32*)&br0[2 * p];
    #pragma unroll
    for (int p = 0; p < 8; p++) bu1[p] = *(const u32*)&br1[2 * p];

    acc0 = __builtin_amdgcn_wmma_f32_16x16x32_bf16(
        false, av, false, as_v16bf(bu0), (short)0, acc0, false, false);
    acc1 = __builtin_amdgcn_wmma_f32_16x16x32_bf16(
        false, av, false, as_v16bf(bu1), (short)0, acc1, false, false);
    __syncthreads();
  }

  // epilogue: D lane layout: N = lane%16, VGPR r -> M = r + 8*(lane/16)
  #pragma unroll
  for (int c = 0; c < 2; c++) {
    v8f acc = c ? acc1 : acc0;
    int col = col0 + cw + c * 16 + bn;
    #pragma unroll
    for (int r = 0; r < 8; r++) {
      int row = row0 + rw + half * 8 + r;
      float v = apply_op<OP>(acc[r]);
      size_t idx = (size_t)row * ldd + col;
      if (EM == EM_BF16)     Dh[idx] = f2bf(v);
      else if (EM == EM_ACC) D[idx] += v;
      else                   D[idx]  = v;
    }
  }
}

// ---------------------------------------------------------------------------
// Elementwise / recurrence kernels
// ---------------------------------------------------------------------------
__global__ void f32_to_bf16_kernel(const float* __restrict__ in,
                                   u16* __restrict__ out, size_t n) {
  size_t i = (size_t)blockIdx.x * blockDim.x + threadIdx.x;
  if (i < n) out[i] = f2bf(in[i]);
}

// column-padded conversion: out[r, 0:srcCols) = bf16(in), out[r, srcCols:) = 0
__global__ void f32_to_bf16_pad_kernel(const float* __restrict__ in,
                                       u16* __restrict__ out,
                                       int srcCols, int dstCols, size_t n) {
  size_t i = (size_t)blockIdx.x * blockDim.x + threadIdx.x;
  if (i >= n) return;
  int c = (int)(i % dstCols);
  size_t r = i / dstCols;
  out[i] = (c < srcCols) ? f2bf(in[r * (size_t)srcCols + c]) : (u16)0;
}

__global__ void posenc_kernel(const float* __restrict__ x,
                              float* __restrict__ hf, float* __restrict__ hb,
                              int B, int T, int C) {
  size_t i = (size_t)blockIdx.x * blockDim.x + threadIdx.x;
  size_t n = (size_t)B * T * C;
  if (i >= n) return;
  int c = (int)(i % C);
  int t = (int)((i / C) % T);
  int b = (int)(i / ((size_t)C * T));
  float e   = -logf(10000.f) * (float)(2 * (c >> 1)) / (float)C;
  float ang = (float)t * expf(e);
  float pe  = (c & 1) ? cosf(ang) : sinf(ang);
  float v   = x[i] + pe;
  hf[i] = v;
  hb[((size_t)b * T + (T - 1 - t)) * C + c] = v;   // time-flipped copy
}

__global__ __launch_bounds__(256) void layernorm_kernel(
    const float* __restrict__ X, float* __restrict__ Y,
    const float* __restrict__ w, const float* __restrict__ bias, int C) {
  __shared__ float red[256];
  int tid = threadIdx.x;
  size_t base = (size_t)blockIdx.x * C;          // C == 512
  float a  = X[base + tid];
  float b2 = X[base + tid + 256];
  red[tid] = a + b2; __syncthreads();
  for (int s = 128; s; s >>= 1) { if (tid < s) red[tid] += red[tid + s]; __syncthreads(); }
  float mu = red[0] / (float)C; __syncthreads();
  float da = a - mu, db = b2 - mu;
  red[tid] = da * da + db * db; __syncthreads();
  for (int s = 128; s; s >>= 1) { if (tid < s) red[tid] += red[tid + s]; __syncthreads(); }
  float rstd = rsqrtf(red[0] / (float)C + 1e-5f); __syncthreads();
  Y[base + tid]       = da * rstd * w[tid]       + bias[tid];
  Y[base + tid + 256] = db * rstd * w[tid + 256] + bias[tid + 256];
}

// out = bf16( xn + (shift(xn)-xn) * coef[c] )
__global__ void mixshift_kernel(const float* __restrict__ XN,
                                const float* __restrict__ coef,
                                u16* __restrict__ out, int T, int C, size_t n) {
  size_t i = (size_t)blockIdx.x * blockDim.x + threadIdx.x;
  if (i >= n) return;
  int c = (int)(i % C);
  int t = (int)((i / C) % T);
  float xv = XN[i];
  float prev = t ? XN[i - C] : 0.f;
  out[i] = f2bf(xv + (prev - xv) * coef[c]);
}

// mix_f = bf16( xn + sx*(maa_wkvrg[f] + mx_f) ), f = 0..4 (w,k,v,r,g)
__global__ void combine5_kernel(const float* __restrict__ XN,
                                const float* __restrict__ MXO,
                                const float* __restrict__ maa,
                                u16* __restrict__ out, int T, int C, size_t MC) {
  size_t i = (size_t)blockIdx.x * blockDim.x + threadIdx.x;
  if (i >= MC) return;
  int c = (int)(i % C);
  int t = (int)((i / C) % T);
  float xv = XN[i];
  float prev = t ? XN[i - C] : 0.f;
  float sx = prev - xv;
  #pragma unroll
  for (int f = 0; f < 5; f++) {
    float v = xv + sx * (maa[f * C + c] + MXO[(size_t)f * MC + i]);
    out[(size_t)f * MC + i] = f2bf(v);
  }
}

// in-place: W = exp(-exp(time_decay[c] + W))
__global__ void wdecay_kernel(float* __restrict__ W,
                              const float* __restrict__ td, int C, size_t n) {
  size_t i = (size_t)blockIdx.x * blockDim.x + threadIdx.x;
  if (i >= n) return;
  int c = (int)(i % C);
  W[i] = expf(-expf(td[c] + W[i]));
}

// WKV6 linear-attention scan: one 64-thread block per (b,h); thread j owns
// value-column j; state column S[0..63] lives in registers.
__global__ __launch_bounds__(64) void wkv_scan_kernel(
    const float* __restrict__ R, const float* __restrict__ Kk,
    const float* __restrict__ V, const float* __restrict__ Wd,
    const float* __restrict__ U, float* __restrict__ O,
    int T, int C, int H) {
  __shared__ float rb[64], kbuf[64], wb[64], ub[64];
  const int tid = threadIdx.x;
  const int b = blockIdx.x / H;
  const int h = blockIdx.x % H;
  ub[tid] = U[h * 64 + tid];
  __syncthreads();
  float S[64];
  #pragma unroll
  for (int i = 0; i < 64; i++) S[i] = 0.f;
  for (int t = 0; t < T; t++) {
    size_t base = ((size_t)b * T + t) * C + h * 64;
    rb[tid]   = R[base + tid];
    kbuf[tid] = Kk[base + tid];
    wb[tid]   = Wd[base + tid];
    __syncthreads();
    float vj = V[base + tid];
    float o = 0.f;
    #pragma unroll
    for (int i = 0; i < 64; i++) {
      float kv = kbuf[i] * vj;
      o += rb[i] * (S[i] + ub[i] * kv);
      S[i] = wb[i] * S[i] + kv;
    }
    O[base + tid] = o;
    __syncthreads();
  }
}

// per-head GroupNorm fused with silu-gate multiply and bf16 conversion
__global__ __launch_bounds__(64) void groupnorm_kernel(
    const float* __restrict__ O, const float* __restrict__ G,
    const float* __restrict__ w, const float* __restrict__ bias,
    u16* __restrict__ OG, int C, int H) {
  __shared__ float red[64];
  int tid = threadIdx.x;
  int h = blockIdx.x % H;
  size_t bt = blockIdx.x / H;
  size_t idx = bt * C + h * 64 + tid;
  float v = O[idx];
  red[tid] = v; __syncthreads();
  for (int s = 32; s; s >>= 1) { if (tid < s) red[tid] += red[tid + s]; __syncthreads(); }
  float mu = red[0] / 64.f; __syncthreads();
  float d = v - mu;
  red[tid] = d * d; __syncthreads();
  for (int s = 32; s; s >>= 1) { if (tid < s) red[tid] += red[tid + s]; __syncthreads(); }
  float var = red[0] / 64.f; __syncthreads();
  float y = d * rsqrtf(var + 64e-5f) * w[h * 64 + tid] + bias[h * 64 + tid];
  OG[idx] = f2bf(y * G[idx]);
}

// h += sigmoid(r) * kv   (RR already sigmoided by GEMM epilogue)
__global__ void rkv_update_kernel(float* __restrict__ h,
                                  const float* __restrict__ RR,
                                  const float* __restrict__ KV, size_t n) {
  size_t i = (size_t)blockIdx.x * blockDim.x + threadIdx.x;
  if (i < n) h[i] += RR[i] * KV[i];
}

// out = (hf + flip_T(hb)) * 0.5 * mask(t < len[b])
__global__ void finalize_kernel(const float* __restrict__ hf,
                                const float* __restrict__ hb,
                                const int* __restrict__ len,
                                float* __restrict__ out, int B, int T, int C) {
  size_t i = (size_t)blockIdx.x * blockDim.x + threadIdx.x;
  size_t n = (size_t)B * T * C;
  if (i >= n) return;
  int c = (int)(i % C);
  int t = (int)((i / C) % T);
  int b = (int)(i / ((size_t)C * T));
  size_t ib = ((size_t)b * T + (T - 1 - t)) * C + c;
  float m = (t < len[b]) ? 0.5f : 0.f;
  out[i] = (hf[i] + hb[ib]) * m;
}

// ---------------------------------------------------------------------------
// Host orchestration
// ---------------------------------------------------------------------------
extern "C" void kernel_launch(void* const* d_in, const int* in_sizes, int n_in,
                              void* d_out, int out_size, void* d_ws, size_t ws_size,
                              hipStream_t stream) {
  (void)in_sizes; (void)n_in; (void)out_size; (void)ws_size;
  const int Bv = 4, Tv = 1024, Cv = 512, Hh = 8, L2 = 4, Fv = 1792;
  const int NP = 192;                       // padded 5*DMIX (160 -> 192)
  const size_t MC = (size_t)Bv * Tv * Cv;   // 2,097,152
  const int Mrows = Bv * Tv;                // 4096

  const float* x          = (const float*)d_in[0];
  const int*   lengths    = (const int*)d_in[1];
  const float* ln0_w      = (const float*)d_in[2];
  const float* ln0_b      = (const float*)d_in[3];
  const float* ln1_w      = (const float*)d_in[4];
  const float* ln1_b      = (const float*)d_in[5];
  const float* ln2_w      = (const float*)d_in[6];
  const float* ln2_b      = (const float*)d_in[7];
  const float* maa_x      = (const float*)d_in[8];
  const float* maa_wkvrg  = (const float*)d_in[9];
  const float* tm_w1      = (const float*)d_in[10];
  const float* tm_w2      = (const float*)d_in[11];
  const float* td_w1      = (const float*)d_in[12];
  const float* td_w2      = (const float*)d_in[13];
  const float* time_decay = (const float*)d_in[14];
  const float* time_faaaa = (const float*)d_in[15];
  const float* Wr         = (const float*)d_in[16];
  const float* Wk         = (const float*)d_in[17];
  const float* Wv         = (const float*)d_in[18];
  const float* Wg         = (const float*)d_in[19];
  const float* Wo         = (const float*)d_in[20];
  const float* lnx_w      = (const float*)d_in[21];
  const float* lnx_b      = (const float*)d_in[22];
  const float* cm_maa_k   = (const float*)d_in[23];
  const float* cm_maa_r   = (const float*)d_in[24];
  const float* cWk        = (const float*)d_in[25];
  const float* cWv        = (const float*)d_in[26];
  const float* cWr        = (const float*)d_in[27];

  char* wsb = (char*)d_ws;
  size_t off = 0;
  auto alloc = [&](size_t bytes) -> void* {
    void* p = wsb + off;
    off = (off + bytes + 255) & ~(size_t)255;
    return p;
  };

  // bf16 weight copies (re-converted every call; deterministic)
  u16* Wr_h   = (u16*)alloc((size_t)L2 * Cv * Cv * 2);
  u16* Wk_h   = (u16*)alloc((size_t)L2 * Cv * Cv * 2);
  u16* Wv_h   = (u16*)alloc((size_t)L2 * Cv * Cv * 2);
  u16* Wg_h   = (u16*)alloc((size_t)L2 * Cv * Cv * 2);
  u16* Wo_h   = (u16*)alloc((size_t)L2 * Cv * Cv * 2);
  u16* cWr_h  = (u16*)alloc((size_t)L2 * Cv * Cv * 2);
  u16* tmw1_h = (u16*)alloc((size_t)L2 * Cv * NP * 2);    // padded to 192
  u16* tmw2_h = (u16*)alloc((size_t)L2 * 5 * 32 * Cv * 2);
  u16* tdw1_h = (u16*)alloc((size_t)L2 * Cv * 64 * 2);
  u16* tdw2_h = (u16*)alloc((size_t)L2 * 64 * Cv * 2);
  u16* cWk_h  = (u16*)alloc((size_t)L2 * Cv * Fv * 2);
  u16* cWv_h  = (u16*)alloc((size_t)L2 * Fv * Cv * 2);

  // activations
  float* hF     = (float*)alloc(MC * 4);
  float* hB     = (float*)alloc(MC * 4);
  float* XN     = (float*)alloc(MC * 4);
  u16*   MIXPRE = (u16*)  alloc(MC * 2);
  u16*   MX     = (u16*)  alloc((size_t)Mrows * NP * 2);  // padded to 192
  float* MXO    = (float*)alloc(5 * MC * 4);
  u16*   MIX5   = (u16*)  alloc(5 * MC * 2);
  float* Rb     = (float*)alloc(MC * 4);
  float* Kb     = (float*)alloc(MC * 4);
  float* Vb     = (float*)alloc(MC * 4);
  float* Wb     = (float*)alloc(MC * 4);
  float* Gb     = (float*)alloc(MC * 4);
  float* Ob     = (float*)alloc(MC * 4);
  u16*   DT     = (u16*)  alloc((size_t)Mrows * 64 * 2);
  u16*   OG     = (u16*)  alloc(MC * 2);
  u16*   KK     = (u16*)  alloc((size_t)Mrows * Fv * 2);
  float* KV     = (float*)alloc(MC * 4);
  float* RR     = (float*)alloc(MC * 4);

  auto conv = [&](const float* src, u16* dst, size_t n) {
    f32_to_bf16_kernel<<<(unsigned)((n + 255) / 256), 256, 0, stream>>>(src, dst, n);
  };
  auto gemm = [&](int op, int em, const u16* A, const u16* Bm, float* D, u16* Dh,
                  int M, int Nn, int K, int lda, int ldb, int ldd) {
    dim3 grid(Nn / 64, M / 64);
    #define GCASE(OPV, EMV) \
      gemm_bf16_kernel<OPV, EMV><<<grid, 256, 0, stream>>>(A, Bm, D, Dh, M, Nn, K, lda, ldb, ldd)
    if      (op == OP_NONE    && em == EM_F32)  GCASE(OP_NONE,    EM_F32);
    else if (op == OP_NONE    && em == EM_ACC)  GCASE(OP_NONE,    EM_ACC);
    else if (op == OP_TANH    && em == EM_BF16) GCASE(OP_TANH,    EM_BF16);
    else if (op == OP_SILU    && em == EM_F32)  GCASE(OP_SILU,    EM_F32);
    else if (op == OP_SIGMOID && em == EM_F32)  GCASE(OP_SIGMOID, EM_F32);
    else if (op == OP_RELUSQ  && em == EM_BF16) GCASE(OP_RELUSQ,  EM_BF16);
    #undef GCASE
  };
  const unsigned EB = (unsigned)((MC + 255) / 256);

  // ---- weight conversion (once per launch) ----
  conv(Wr,   Wr_h,   (size_t)L2 * Cv * Cv);
  conv(Wk,   Wk_h,   (size_t)L2 * Cv * Cv);
  conv(Wv,   Wv_h,   (size_t)L2 * Cv * Cv);
  conv(Wg,   Wg_h,   (size_t)L2 * Cv * Cv);
  conv(Wo,   Wo_h,   (size_t)L2 * Cv * Cv);
  conv(cWr,  cWr_h,  (size_t)L2 * Cv * Cv);
  { // tm_w1: pad 160 -> 192 columns with zeros
    size_t n = (size_t)L2 * Cv * NP;
    f32_to_bf16_pad_kernel<<<(unsigned)((n + 255) / 256), 256, 0, stream>>>(
        tm_w1, tmw1_h, 160, NP, n);
  }
  conv(tm_w2, tmw2_h, (size_t)L2 * 5 * 32 * Cv);
  conv(td_w1, tdw1_h, (size_t)L2 * Cv * 64);
  conv(td_w2, tdw2_h, (size_t)L2 * 64 * Cv);
  conv(cWk,  cWk_h,  (size_t)L2 * Cv * Fv);
  conv(cWv,  cWv_h,  (size_t)L2 * Fv * Cv);

  // ---- positional encoding + forward / time-flipped backward streams ----
  posenc_kernel<<<EB, 256, 0, stream>>>(x, hF, hB, Bv, Tv, Cv);

  for (int d = 0; d < 2; d++) {
    float* h = d ? hB : hF;
    layernorm_kernel<<<Mrows, 256, 0, stream>>>(h, h, ln0_w + d * Cv, ln0_b + d * Cv, Cv);

    for (int il = 0; il < 2; il++) {
      const int li = d * 2 + il;

      // ===================== time-mix =====================
      layernorm_kernel<<<Mrows, 256, 0, stream>>>(h, XN, ln1_w + li * Cv, ln1_b + li * Cv, Cv);
      mixshift_kernel<<<EB, 256, 0, stream>>>(XN, maa_x + li * Cv, MIXPRE, Tv, Cv, MC);
      // mx = tanh(mixpre @ tm_w1)  [4096,192(pad)] bf16
      gemm(OP_TANH, EM_BF16, MIXPRE, tmw1_h + (size_t)li * Cv * NP, nullptr, MX,
           Mrows, NP, Cv, Cv, NP, NP);
      // per-channel LoRA up-projection (K=32 each)
      for (int f = 0; f < 5; f++)
        gemm(OP_NONE, EM_F32, MX + f * 32, tmw2_h + ((size_t)li * 5 + f) * 32 * Cv,
             MXO + (size_t)f * MC, nullptr, Mrows, Cv, 32, NP, Cv, Cv);
      combine5_kernel<<<EB, 256, 0, stream>>>(XN, MXO, maa_wkvrg + (size_t)li * 5 * Cv,
                                              MIX5, Tv, Cv, MC);
      // big projections (f order: 0=w 1=k 2=v 3=r 4=g)
      gemm(OP_NONE, EM_F32, MIX5 + 3 * MC, Wr_h + (size_t)li * Cv * Cv, Rb, nullptr,
           Mrows, Cv, Cv, Cv, Cv, Cv);
      gemm(OP_NONE, EM_F32, MIX5 + 1 * MC, Wk_h + (size_t)li * Cv * Cv, Kb, nullptr,
           Mrows, Cv, Cv, Cv, Cv, Cv);
      gemm(OP_NONE, EM_F32, MIX5 + 2 * MC, Wv_h + (size_t)li * Cv * Cv, Vb, nullptr,
           Mrows, Cv, Cv, Cv, Cv, Cv);
      gemm(OP_SILU, EM_F32, MIX5 + 4 * MC, Wg_h + (size_t)li * Cv * Cv, Gb, nullptr,
           Mrows, Cv, Cv, Cv, Cv, Cv);
      // decay LoRA: tanh(mix_w @ td_w1) @ td_w2; then exp(-exp(td + .))
      gemm(OP_TANH, EM_BF16, MIX5 + 0 * MC, tdw1_h + (size_t)li * Cv * 64, nullptr, DT,
           Mrows, 64, Cv, Cv, 64, 64);
      gemm(OP_NONE, EM_F32, DT, tdw2_h + (size_t)li * 64 * Cv, Wb, nullptr,
           Mrows, Cv, 64, 64, Cv, Cv);
      wdecay_kernel<<<EB, 256, 0, stream>>>(Wb, time_decay + li * Cv, Cv, MC);
      // WKV recurrence
      wkv_scan_kernel<<<Bv * Hh, 64, 0, stream>>>(Rb, Kb, Vb, Wb,
                                                  time_faaaa + (size_t)li * Hh * 64,
                                                  Ob, Tv, Cv, Hh);
      // per-head GroupNorm * silu gate -> bf16, then h += og @ Wo
      groupnorm_kernel<<<Mrows * Hh, 64, 0, stream>>>(Ob, Gb, lnx_w + li * Cv,
                                                      lnx_b + li * Cv, OG, Cv, Hh);
      gemm(OP_NONE, EM_ACC, OG, Wo_h + (size_t)li * Cv * Cv, h, nullptr,
           Mrows, Cv, Cv, Cv, Cv, Cv);

      // ===================== channel-mix =====================
      layernorm_kernel<<<Mrows, 256, 0, stream>>>(h, XN, ln2_w + li * Cv, ln2_b + li * Cv, Cv);
      mixshift_kernel<<<EB, 256, 0, stream>>>(XN, cm_maa_k + li * Cv, MIXPRE, Tv, Cv, MC);
      gemm(OP_RELUSQ, EM_BF16, MIXPRE, cWk_h + (size_t)li * Cv * Fv, nullptr, KK,
           Mrows, Fv, Cv, Cv, Fv, Fv);
      gemm(OP_NONE, EM_F32, KK, cWv_h + (size_t)li * Fv * Cv, KV, nullptr,
           Mrows, Cv, Fv, Fv, Cv, Cv);
      mixshift_kernel<<<EB, 256, 0, stream>>>(XN, cm_maa_r + li * Cv, MIXPRE, Tv, Cv, MC);
      gemm(OP_SIGMOID, EM_F32, MIXPRE, cWr_h + (size_t)li * Cv * Cv, RR, nullptr,
           Mrows, Cv, Cv, Cv, Cv, Cv);
      rkv_update_kernel<<<EB, 256, 0, stream>>>(h, RR, KV, MC);
    }
  }

  finalize_kernel<<<EB, 256, 0, stream>>>(hF, hB, lengths, (float*)d_out, Bv, Tv, Cv);
}
